// Longformer_base_Net_73607149519685
// MI455X (gfx1250) — compile-verified, matrix operand order checked by reading, not run on hardware
//
#include <hip/hip_runtime.h>
#include <hip/hip_bf16.h>

typedef _Float16 f16;
typedef __attribute__((ext_vector_type(16))) _Float16 v16h;
typedef __attribute__((ext_vector_type(8)))  _Float16 v8h;
typedef __attribute__((ext_vector_type(8)))  float    v8f;
typedef __attribute__((ext_vector_type(4)))  int      v4i;

#define LDA 40   // padded LDS stride (f16 units); 80B rows keep 16B alignment

// CDNA5 async global->LDS copy path (ASYNCcnt), guarded: falls back cleanly.
#if __has_builtin(__builtin_amdgcn_global_load_async_to_lds_b128) && \
    __has_builtin(__builtin_amdgcn_s_wait_asynccnt)
#define USE_ASYNC_LDS 1
#else
#define USE_ASYNC_LDS 0
#endif

#if USE_ASYNC_LDS
typedef __attribute__((address_space(3))) v4i as3_v4i;

__device__ __forceinline__ void async_copy16(const f16* g, f16* l)
{
    // param1: generic v4i* (global source); param2: AS(3) v4i* (LDS dest);
    // imm offset, cpol
    __builtin_amdgcn_global_load_async_to_lds_b128((v4i*)g, (as3_v4i*)l, 0, 0);
}
#endif

// CDNA5 Tensor Data Mover path (TENSORcnt), guarded likewise.
#if __has_builtin(__builtin_amdgcn_tensor_load_to_lds) && \
    __has_builtin(__builtin_amdgcn_s_wait_tensorcnt)
#define USE_TDM 1
#else
#define USE_TDM 0
#endif

#if USE_TDM
typedef __attribute__((ext_vector_type(4))) unsigned int v4u;
typedef __attribute__((ext_vector_type(8))) int          v8i;

__device__ __forceinline__ void tdm_load(v4u g0, v8i g1)
{
    v4i z4 = {};
#if __has_include(<hip/amd_detail/amd_gfx1250_TDM.h>)
    v8i z8 = {};
    __builtin_amdgcn_tensor_load_to_lds(g0, g1, z4, z4, z8, 0);   // 6-arg toolchain
#else
    __builtin_amdgcn_tensor_load_to_lds(g0, g1, z4, z4, 0);       // 5-arg toolchain
#endif
}
#endif

// shared fragment-load + 8x wmma step for one 128x128x32 tile slice
__device__ __forceinline__ void wmma_tile_step(
    const f16* Asb, const f16* Bsb, int wm, int wn, int hf, int l16,
    v8f (&acc)[2][4])
{
    v16h a[2], b[4];
#pragma unroll
    for (int i = 0; i < 2; ++i) {
        int r = wm * 32 + i * 16 + l16;
        union { v16h v; v8h h[2]; } t;
        t.h[0] = *(const v8h*)&Asb[r * LDA + hf * 8];
        t.h[1] = *(const v8h*)&Asb[r * LDA + hf * 8 + 16];
        a[i] = t.v;
    }
#pragma unroll
    for (int j = 0; j < 4; ++j) {
        int r = wn * 64 + j * 16 + l16;
        union { v16h v; v8h h[2]; } t;
        t.h[0] = *(const v8h*)&Bsb[r * LDA + hf * 8];
        t.h[1] = *(const v8h*)&Bsb[r * LDA + hf * 8 + 16];
        b[j] = t.v;
    }
#pragma unroll
    for (int i = 0; i < 2; ++i)
#pragma unroll
        for (int j = 0; j < 4; ++j)
            acc[i][j] = __builtin_amdgcn_wmma_f32_16x16x32_f16(
                false, a[i], false, b[j], (short)0, acc[i][j], false, false);
}

// ---------------------------------------------------------------------------
// Tiled WMMA GEMM:  C[M,N] = A[M,K] (f16) x Bt[N,K]^T (f16) + bias, + epilogue
// Block: 256 threads (8 waves), tile 128x128, BK=32 (one wmma K step).
// EPI: 0 = f16 out, 1 = f16 out *0.125 (Q), 2 = f16 gelu(out), 3 = f32 out + residual
// ---------------------------------------------------------------------------
template <int EPI>
__global__ __launch_bounds__(256) void gemm_wmma_kernel(
    const f16* __restrict__ A, const f16* __restrict__ Bt,
    const float* __restrict__ bias,
    f16* __restrict__ outH, float* __restrict__ outF,
    const float* __restrict__ res, int M, int N, int K)
{
    const int tid  = threadIdx.x;
    const int lane = tid & 31;
    const int wid  = tid >> 5;
    const int wm   = wid & 3;        // wave row    (32 rows each -> 128)
    const int wn   = wid >> 2;       // wave column (64 cols each -> 128)
    const int hf   = lane >> 4;      // lane half (0: lanes 0-15, 1: lanes 16-31)
    const int l16  = lane & 15;
    const int m0   = blockIdx.y * 128;
    const int n0   = blockIdx.x * 128;

    // per-thread 16B chunk assignment (2 chunks each for A and B tiles)
    const int r0 = tid >> 2,           kc0 = (tid & 3) * 8;
    const int r1 = (tid + 256) >> 2,   kc1 = (tid & 3) * 8;  // (tid+256)&3 == tid&3

    v8f acc[2][4] = {};

#if USE_ASYNC_LDS
    // double-buffered LDS fed by async DMA (ASYNCcnt); 4 async ops/thread/stage
    __shared__ __attribute__((aligned(16))) f16 As[2][128 * LDA];
    __shared__ __attribute__((aligned(16))) f16 Bs[2][128 * LDA];

    auto stage = [&](int buf, int k0) {
        async_copy16(&A [(size_t)(m0 + r0) * K + k0 + kc0], &As[buf][r0 * LDA + kc0]);
        async_copy16(&A [(size_t)(m0 + r1) * K + k0 + kc1], &As[buf][r1 * LDA + kc1]);
        async_copy16(&Bt[(size_t)(n0 + r0) * K + k0 + kc0], &Bs[buf][r0 * LDA + kc0]);
        async_copy16(&Bt[(size_t)(n0 + r1) * K + k0 + kc1], &Bs[buf][r1 * LDA + kc1]);
    };
    stage(0, 0);
    int buf = 0;
    for (int k0 = 0; k0 < K; k0 += 32, buf ^= 1) {
        if (k0 + 32 < K) {
            stage(buf ^ 1, k0 + 32);                 // prefetch next slice via DMA
            __builtin_amdgcn_s_wait_asynccnt(4);     // current slice's 4 ops done
        } else {
            __builtin_amdgcn_s_wait_asynccnt(0);
        }
        __syncthreads();
        wmma_tile_step(&As[buf][0], &Bs[buf][0], wm, wn, hf, l16, acc);
        __syncthreads();
    }
#else
    // fallback: register-staged single buffer + cache prefetch of next slice
    __shared__ __attribute__((aligned(16))) f16 As[1][128 * LDA];
    __shared__ __attribute__((aligned(16))) f16 Bs[1][128 * LDA];

    for (int k0 = 0; k0 < K; k0 += 32) {
        *(v8h*)&As[0][r0 * LDA + kc0] = *(const v8h*)&A [(size_t)(m0 + r0) * K + k0 + kc0];
        *(v8h*)&As[0][r1 * LDA + kc1] = *(const v8h*)&A [(size_t)(m0 + r1) * K + k0 + kc1];
        *(v8h*)&Bs[0][r0 * LDA + kc0] = *(const v8h*)&Bt[(size_t)(n0 + r0) * K + k0 + kc0];
        *(v8h*)&Bs[0][r1 * LDA + kc1] = *(const v8h*)&Bt[(size_t)(n0 + r1) * K + k0 + kc1];
        if (k0 + 32 < K)
            __builtin_prefetch(&A[(size_t)(m0 + (tid >> 1)) * K + k0 + 32], 0, 1);
        __syncthreads();
        wmma_tile_step(&As[0][0], &Bs[0][0], wm, wn, hf, l16, acc);
        __syncthreads();
    }
#endif

    // epilogue: C/D layout -> VGPR r holds M = r (lanes 0-15) / M = 8+r (lanes 16-31)
#pragma unroll
    for (int i = 0; i < 2; ++i) {
#pragma unroll
        for (int j = 0; j < 4; ++j) {
            int col  = n0 + wn * 64 + j * 16 + l16;
            float bv = bias ? bias[col] : 0.0f;
#pragma unroll
            for (int r = 0; r < 8; ++r) {
                int row    = m0 + wm * 32 + i * 16 + hf * 8 + r;
                float v    = acc[i][j][r] + bv;
                size_t idx = (size_t)row * N + col;
                if (EPI == 0) {
                    outH[idx] = (f16)v;
                } else if (EPI == 1) {
                    outH[idx] = (f16)(v * 0.125f);
                } else if (EPI == 2) {
                    outH[idx] = (f16)(0.5f * v * (1.0f + erff(v * 0.70710678f)));
                } else {
                    outF[idx] = v + res[idx];
                }
            }
        }
    }
}

// ---------------------------------------------------------------------------
// conv0 as implicit GEMM on WMMA: M = 256*37 = 9472 (= 74*128), N = 128,
// K = 8*768 = 6144 ordered k = kk*768 + c so each 32-wide K slice is
// contiguous in x16. B (weight) tile loaded by the Tensor Data Mover when
// available: 2D tensor, tile 128x32 f16, LDS pad 4 DWORDs every 16 DWORDs
// (reproduces the LDA=40 padded stride). Fused bias+BN+ReLU epilogue.
// ---------------------------------------------------------------------------
__global__ __launch_bounds__(256) void conv0_wmma_kernel(
    const f16* __restrict__ X,   // x16: [20736, 768], row = b*81 + s
    const f16* __restrict__ Wt,  // [128, 6144], k = kk*768 + c
    const float* __restrict__ cb, const float* __restrict__ g,
    const float* __restrict__ bb, const float* __restrict__ bm,
    const float* __restrict__ bv, float* __restrict__ out)  // [256,128,37]
{
    __shared__ __attribute__((aligned(16))) f16 As[128 * LDA];
    __shared__ __attribute__((aligned(16))) f16 Bs[128 * LDA];

    const int tid  = threadIdx.x;
    const int lane = tid & 31;
    const int wid  = tid >> 5;
    const int wm   = wid & 3;
    const int wn   = wid >> 2;
    const int hf   = lane >> 4;
    const int l16  = lane & 15;
    const int m0   = blockIdx.y * 128;   // 74 blocks, N block = 0 only

    const int r0 = tid >> 2,          kc0 = (tid & 3) * 8;
    const int r1 = (tid + 256) >> 2,  kc1 = (tid & 3) * 8;
    const int b0 = (m0 + r0) / 37, t0 = (m0 + r0) % 37;
    const int b1 = (m0 + r1) / 37, t1 = (m0 + r1) % 37;

    v8f acc[2][4] = {};

#if USE_TDM
    const unsigned int ldsB = (unsigned int)(unsigned long long)
        (__attribute__((address_space(3))) void*)&Bs[0];
#endif

    for (int k0 = 0; k0 < 6144; k0 += 32) {
        const int kk = k0 / 768;        // conv tap (0..7)
        const int c0 = k0 % 768;        // channel base
        *(v8h*)&As[r0 * LDA + kc0] =
            *(const v8h*)&X[((size_t)b0 * 81 + 2 * t0 + kk) * 768 + c0 + kc0];
        *(v8h*)&As[r1 * LDA + kc1] =
            *(const v8h*)&X[((size_t)b1 * 81 + 2 * t1 + kk) * 768 + c0 + kc1];
#if USE_TDM
        if (wid == 0) {   // wave-uniform: TDM issues once, ignores EXEC
            unsigned long long ga = (unsigned long long)(size_t)&Wt[k0];
            v4u g0;
            g0.x = 1u;                                   // count=1, user mode
            g0.y = ldsB;                                 // lds_addr
            g0.z = (unsigned int)ga;                     // global_addr[31:0]
            g0.w = (unsigned int)((ga >> 32) & 0x01FFFFFFull) | (2u << 30); // [56:32]|type=2
            v8i g1;
            g1[0] = (int)((1u << 16)        // data_size = 2 bytes
                        | (1u << 20)        // pad_enable
                        | (3u << 22)        // pad_interval: 16 DWORDs
                        | (3u << 25));      // pad_amount: 4 DWORDs
            g1[1] = (int)((6144u & 0xFFFFu) << 16);      // tensor_dim0 lo16 @ [31:16]
            g1[2] = (int)((6144u >> 16) | (128u << 16)); // dim0 hi16 | tensor_dim1 lo16
            g1[3] = (int)(32u << 16);                    // dim1 hi16 | tile_dim0 = 32
            g1[4] = 128;                                 // tile_dim1 = 128, tile_dim2 = 0
            g1[5] = 6144;                                // tensor_dim0_stride[31:0]
            g1[6] = 0;
            g1[7] = 0;
            tdm_load(g0, g1);
            __builtin_amdgcn_s_wait_tensorcnt(0);
        }
#else
        *(v8h*)&Bs[r0 * LDA + kc0] = *(const v8h*)&Wt[(size_t)r0 * 6144 + k0 + kc0];
        *(v8h*)&Bs[r1 * LDA + kc1] = *(const v8h*)&Wt[(size_t)r1 * 6144 + k0 + kc1];
#endif
        __syncthreads();
        wmma_tile_step(As, Bs, wm, wn, hf, l16, acc);
        __syncthreads();
    }

#pragma unroll
    for (int i = 0; i < 2; ++i) {
#pragma unroll
        for (int j = 0; j < 4; ++j) {
            int oc = wn * 64 + j * 16 + l16;
            float scale = g[oc] * rsqrtf(bv[oc] + 1e-5f);
            float shift = bb[oc] - bm[oc] * scale;
            float bias0 = cb[oc];
#pragma unroll
            for (int r = 0; r < 8; ++r) {
                int row = m0 + wm * 32 + i * 16 + hf * 8 + r;
                int b = row / 37, t = row % 37;
                float v = (acc[i][j][r] + bias0) * scale + shift;
                out[((size_t)b * 128 + oc) * 37 + t] = fmaxf(v, 0.0f);
            }
        }
    }
}

// conv0 weight transform: [128][768][8] fp32 -> f16 [128][k = kk*768 + c]
__global__ void convert_conv0_kernel(const float* __restrict__ W, f16* __restrict__ Wt)
{
    int idx = blockIdx.x * 256 + threadIdx.x;   // 128*6144
    if (idx >= 128 * 6144) return;
    int oc = idx / 6144, k = idx % 6144;
    int kk = k / 768, c = k % 768;
    Wt[idx] = (f16)W[(size_t)oc * 6144 + c * 8 + kk];
}

// ---------------------------------------------------------------------------
// Attention: one workgroup (128 thr / 4 waves) per (batch, head).
// S = Q K^T (wmma, K=64), f32 softmax over 81 cols, O = P V (wmma, K=96).
// ---------------------------------------------------------------------------
__global__ __launch_bounds__(128) void attn_kernel(
    const f16* __restrict__ q, const f16* __restrict__ k,
    const f16* __restrict__ v, f16* __restrict__ o)
{
    __shared__ __attribute__((aligned(16))) f16 Qs[96 * 72];   // 96 x 64 (+8 pad)
    __shared__ __attribute__((aligned(16))) f16 Ks[96 * 72];
    __shared__ __attribute__((aligned(16))) f16 Vt[64 * 104];  // V^T: 64 x 96 (+8 pad)
    __shared__ __attribute__((aligned(16))) f16 Sp[96 * 104];  // scores -> probs

    const int tid  = threadIdx.x;
    const int lane = tid & 31;
    const int wid  = tid >> 5;
    const int hf   = lane >> 4;
    const int l16  = lane & 15;
    const int bh   = blockIdx.x;           // b*12 + h
    const int b    = bh / 12, h = bh % 12;
    const size_t base = ((size_t)b * 81) * 768 + (size_t)h * 64;

    for (int idx = tid; idx < 96 * 64; idx += 128) {
        int r = idx >> 6, c = idx & 63;
        f16 qv = (f16)0, kv = (f16)0;
        if (r < 81) {
            qv = q[base + (size_t)r * 768 + c];
            kv = k[base + (size_t)r * 768 + c];
        }
        Qs[r * 72 + c] = qv;
        Ks[r * 72 + c] = kv;
    }
    for (int idx = tid; idx < 64 * 96; idx += 128) {
        int d = idx / 96, s = idx % 96;
        Vt[d * 104 + s] = (s < 81) ? v[base + (size_t)s * 768 + d] : (f16)0;
    }
    __syncthreads();

    for (int t = wid; t < 36; t += 4) {       // S = Q K^T, wave-uniform tiles
        int ti = t / 6, tj = t % 6;
        v8f acc = {};
#pragma unroll
        for (int ks = 0; ks < 2; ++ks) {
            union { v16h v; v8h h[2]; } ta, tb;
            int ra = ti * 16 + l16, rb = tj * 16 + l16;
            int kc = ks * 32 + hf * 8;
            ta.h[0] = *(const v8h*)&Qs[ra * 72 + kc];
            ta.h[1] = *(const v8h*)&Qs[ra * 72 + kc + 16];
            tb.h[0] = *(const v8h*)&Ks[rb * 72 + kc];
            tb.h[1] = *(const v8h*)&Ks[rb * 72 + kc + 16];
            acc = __builtin_amdgcn_wmma_f32_16x16x32_f16(
                false, ta.v, false, tb.v, (short)0, acc, false, false);
        }
#pragma unroll
        for (int r = 0; r < 8; ++r)
            Sp[(ti * 16 + hf * 8 + r) * 104 + tj * 16 + l16] = (f16)acc[r];
    }
    __syncthreads();

    if (tid < 96) {                            // row softmax, zero padding
        if (tid < 81) {
            float mx = -1e30f;
            for (int c = 0; c < 81; ++c) mx = fmaxf(mx, (float)Sp[tid * 104 + c]);
            float sum = 0.0f;
            for (int c = 0; c < 81; ++c) {
                float e = __expf((float)Sp[tid * 104 + c] - mx);
                Sp[tid * 104 + c] = (f16)e;
                sum += e;
            }
            float inv = 1.0f / sum;
            for (int c = 0; c < 81; ++c)
                Sp[tid * 104 + c] = (f16)((float)Sp[tid * 104 + c] * inv);
            for (int c = 81; c < 96; ++c) Sp[tid * 104 + c] = (f16)0;
        } else {
            for (int c = 0; c < 96; ++c) Sp[tid * 104 + c] = (f16)0;
        }
    }
    __syncthreads();

    for (int t = wid; t < 24; t += 4) {        // O = P V
        int ti = t >> 2, nj = t & 3;
        v8f acc = {};
#pragma unroll
        for (int ks = 0; ks < 3; ++ks) {
            union { v16h v; v8h h[2]; } ta, tb;
            int kc = ks * 32 + hf * 8;
            int ra = ti * 16 + l16, rb = nj * 16 + l16;
            ta.h[0] = *(const v8h*)&Sp[ra * 104 + kc];
            ta.h[1] = *(const v8h*)&Sp[ra * 104 + kc + 16];
            tb.h[0] = *(const v8h*)&Vt[rb * 104 + kc];
            tb.h[1] = *(const v8h*)&Vt[rb * 104 + kc + 16];
            acc = __builtin_amdgcn_wmma_f32_16x16x32_f16(
                false, ta.v, false, tb.v, (short)0, acc, false, false);
        }
#pragma unroll
        for (int r = 0; r < 8; ++r) {
            int s = ti * 16 + hf * 8 + r;
            if (s < 81)
                o[base + (size_t)s * 768 + nj * 16 + l16] = (f16)acc[r];
        }
    }
}

// ---------------------------------------------------------------------------
// LayerNorm over D=768: one block per row; writes f32 and f16 copies.
// ---------------------------------------------------------------------------
__global__ __launch_bounds__(256) void layernorm_kernel(
    const float* __restrict__ in, const float* __restrict__ g,
    const float* __restrict__ bb, float* __restrict__ out32,
    f16* __restrict__ out16)
{
    __shared__ float s1[256], s2[256];
    const int row = blockIdx.x, tid = threadIdx.x;
    const size_t rb = (size_t)row * 768;
    float x0 = in[rb + tid], x1 = in[rb + tid + 256], x2 = in[rb + tid + 512];
    s1[tid] = x0 + x1 + x2;
    s2[tid] = x0 * x0 + x1 * x1 + x2 * x2;
    __syncthreads();
    for (int off = 128; off > 0; off >>= 1) {
        if (tid < off) { s1[tid] += s1[tid + off]; s2[tid] += s2[tid + off]; }
        __syncthreads();
    }
    float mean = s1[0] * (1.0f / 768.0f);
    float var  = s2[0] * (1.0f / 768.0f) - mean * mean;
    float inv  = rsqrtf(var + 1e-5f);
#pragma unroll
    for (int i = 0; i < 3; ++i) {
        int c = tid + i * 256;
        float xv = (i == 0) ? x0 : ((i == 1) ? x1 : x2);
        float yv = (xv - mean) * inv * g[c] + bb[c];
        out32[rb + c] = yv;
        out16[rb + c] = (f16)yv;
    }
}

__global__ void embed_kernel(const int* __restrict__ ids,
                             const float* __restrict__ we,
                             const float* __restrict__ pe,
                             float* __restrict__ out)
{
    size_t idx = (size_t)blockIdx.x * 256 + threadIdx.x;  // over M*768
    int tok = (int)(idx / 768), c = (int)(idx % 768);
    int s = tok % 81;
    out[idx] = we[(size_t)ids[tok] * 768 + c] + pe[(size_t)s * 768 + c];
}

// fp32 [K,N] -> f16 transposed [N,K]
__global__ void convert_transpose_kernel(const float* __restrict__ W,
                                         f16* __restrict__ Wt, int K, int N)
{
    size_t idx = (size_t)blockIdx.x * 256 + threadIdx.x;
    if (idx >= (size_t)K * N) return;
    int kk = (int)(idx / N), n = (int)(idx % N);
    Wt[(size_t)n * K + kk] = (f16)W[idx];
}

// Conv1d (VALID) + eval BatchNorm + ReLU (small convs, channel-first input)
__global__ void conv_bn_relu_kernel(
    const float* __restrict__ in, const float* __restrict__ w,
    const float* __restrict__ cb, const float* __restrict__ g,
    const float* __restrict__ bb, const float* __restrict__ bm,
    const float* __restrict__ bv, float* __restrict__ out,
    int Cin, int Lin, int Cout, int Kw, int stride, int Lout)
{
    int idx = blockIdx.x * 256 + threadIdx.x;
    int total = 256 * Cout * Lout;
    if (idx >= total) return;
    int t  = idx % Lout;
    int oc = (idx / Lout) % Cout;
    int bi = idx / (Lout * Cout);
    float acc = 0.0f;
    for (int c = 0; c < Cin; ++c)
        for (int kk = 0; kk < Kw; ++kk)
            acc += in[((size_t)bi * Cin + c) * Lin + t * stride + kk] *
                   w[((size_t)oc * Cin + c) * Kw + kk];
    acc += cb[oc];
    acc = (acc - bm[oc]) * (g[oc] * rsqrtf(bv[oc] + 1e-5f)) + bb[oc];
    out[idx] = fmaxf(acc, 0.0f);
}

__global__ void maxpool3_kernel(const float* __restrict__ in,
                                float* __restrict__ out, int C, int Lin, int Lout)
{
    int idx = blockIdx.x * 256 + threadIdx.x;
    int total = 256 * C * Lout;
    if (idx >= total) return;
    int t = idx % Lout;
    int c = (idx / Lout) % C;
    int bi = idx / (Lout * C);
    const float* p = &in[((size_t)bi * C + c) * Lin + t];
    out[idx] = fmaxf(p[0], fmaxf(p[1], p[2]));
}

__global__ void dense1_kernel(const float* __restrict__ in,
                              const float* __restrict__ w,
                              const float* __restrict__ b,
                              float* __restrict__ out)
{
    int idx = blockIdx.x * 256 + threadIdx.x;  // 256*32
    if (idx >= 256 * 32) return;
    int o = idx % 32, bi = idx / 32;
    float acc = b[o];
    for (int i = 0; i < 96; ++i) acc += in[bi * 96 + i] * w[o * 96 + i];
    out[idx] = fmaxf(acc, 0.0f);
}

__global__ void dense2_kernel(const float* __restrict__ in,
                              const float* __restrict__ w,
                              const float* __restrict__ b,
                              float* __restrict__ out)
{
    int bi = blockIdx.x * 256 + threadIdx.x;
    if (bi >= 256) return;
    float acc = b[0];
    for (int i = 0; i < 32; ++i) acc += in[bi * 32 + i] * w[i];
    out[bi] = 1.0f / (1.0f + __expf(-acc));
}

// ---------------------------------------------------------------------------
extern "C" void kernel_launch(void* const* d_in, const int* in_sizes, int n_in,
                              void* d_out, int out_size, void* d_ws, size_t ws_size,
                              hipStream_t stream)
{
    (void)in_sizes; (void)n_in; (void)out_size; (void)ws_size;
    const size_t M = 20736;  // 256 * 81
    const int Mi = (int)M;

    const int*   input_ids = (const int*)d_in[0];
    const float* word_emb  = (const float*)d_in[1];
    const float* pos_emb   = (const float*)d_in[2];
    const float* emb_ln_g  = (const float*)d_in[3];
    const float* emb_ln_b  = (const float*)d_in[4];
    const float* wq = (const float*)d_in[5];   const float* bq = (const float*)d_in[6];
    const float* wk = (const float*)d_in[7];   const float* bk = (const float*)d_in[8];
    const float* wv = (const float*)d_in[9];   const float* bv = (const float*)d_in[10];
    const float* wo = (const float*)d_in[11];  const float* bo = (const float*)d_in[12];
    const float* ln1_g = (const float*)d_in[13]; const float* ln1_b = (const float*)d_in[14];
    const float* w1 = (const float*)d_in[15];  const float* b1 = (const float*)d_in[16];
    const float* w2 = (const float*)d_in[17];  const float* b2 = (const float*)d_in[18];
    const float* ln2_g = (const float*)d_in[19]; const float* ln2_b = (const float*)d_in[20];
    const float* conv0_w = (const float*)d_in[21]; const float* conv0_b = (const float*)d_in[22];
    const float* bn0_g = (const float*)d_in[23]; const float* bn0_b = (const float*)d_in[24];
    const float* bn0_m = (const float*)d_in[25]; const float* bn0_v = (const float*)d_in[26];
    const float* conv1_w = (const float*)d_in[27]; const float* conv1_b = (const float*)d_in[28];
    const float* bn1_g = (const float*)d_in[29]; const float* bn1_b = (const float*)d_in[30];
    const float* bn1_m = (const float*)d_in[31]; const float* bn1_v = (const float*)d_in[32];
    const float* conv2_w = (const float*)d_in[33]; const float* conv2_b = (const float*)d_in[34];
    const float* bn2_g = (const float*)d_in[35]; const float* bn2_b = (const float*)d_in[36];
    const float* bn2_m = (const float*)d_in[37]; const float* bn2_v = (const float*)d_in[38];
    const float* d1_w = (const float*)d_in[39]; const float* d1_b = (const float*)d_in[40];
    const float* d2_w = (const float*)d_in[41]; const float* d2_b = (const float*)d_in[42];
    float* out = (float*)d_out;

    // workspace carve (bump allocator, 256B aligned)
    char* p = (char*)d_ws;
    auto alloc = [&](size_t bytes) -> char* {
        char* r = p;
        p += (bytes + 255) & ~(size_t)255;
        return r;
    };
    float* x32  = (float*)alloc(M * 768 * 4);
    f16*   x16  = (f16*)  alloc(M * 768 * 2);
    f16*   q16  = (f16*)  alloc(M * 768 * 2);
    f16*   k16  = (f16*)  alloc(M * 768 * 2);
    f16*   v16  = (f16*)  alloc(M * 768 * 2);
    f16*   o16  = (f16*)  alloc(M * 768 * 2);
    float* y32  = (float*)alloc(M * 768 * 4);
    f16*   ff16 = (f16*)  alloc(M * 3072 * 2);
    f16*   wt   = (f16*)  alloc((size_t)3072 * 768 * 2);
    float* c0 = (float*)alloc((size_t)256 * 128 * 37 * 4);
    float* p0 = (float*)alloc((size_t)256 * 128 * 35 * 4);
    float* c1 = (float*)alloc((size_t)256 * 64 * 14 * 4);
    float* p1 = (float*)alloc((size_t)256 * 64 * 12 * 4);
    float* c2 = (float*)alloc((size_t)256 * 32 * 5 * 4);
    float* p2 = (float*)alloc((size_t)256 * 32 * 3 * 4);
    float* fc1 = (float*)alloc((size_t)256 * 32 * 4);

    embed_kernel<<<62208, 256, 0, stream>>>(input_ids, word_emb, pos_emb, y32);
    layernorm_kernel<<<Mi, 256, 0, stream>>>(y32, emb_ln_g, emb_ln_b, x32, x16);

    const dim3 g768(6, 162), g3072(24, 162);
    const int ctD = (768 * 768) / 256;     // 2304 blocks
    const int ctF = (768 * 3072) / 256;    // 9216 blocks

    for (int l = 0; l < 12; ++l) {
        const size_t wD = (size_t)l * 768 * 768;
        const size_t wF = (size_t)l * 768 * 3072;
        convert_transpose_kernel<<<ctD, 256, 0, stream>>>(wq + wD, wt, 768, 768);
        gemm_wmma_kernel<1><<<g768, 256, 0, stream>>>(x16, wt, bq + l * 768, q16, nullptr, nullptr, Mi, 768, 768);
        convert_transpose_kernel<<<ctD, 256, 0, stream>>>(wk + wD, wt, 768, 768);
        gemm_wmma_kernel<0><<<g768, 256, 0, stream>>>(x16, wt, bk + l * 768, k16, nullptr, nullptr, Mi, 768, 768);
        convert_transpose_kernel<<<ctD, 256, 0, stream>>>(wv + wD, wt, 768, 768);
        gemm_wmma_kernel<0><<<g768, 256, 0, stream>>>(x16, wt, bv + l * 768, v16, nullptr, nullptr, Mi, 768, 768);
        attn_kernel<<<3072, 128, 0, stream>>>(q16, k16, v16, o16);
        convert_transpose_kernel<<<ctD, 256, 0, stream>>>(wo + wD, wt, 768, 768);
        gemm_wmma_kernel<3><<<g768, 256, 0, stream>>>(o16, wt, bo + l * 768, nullptr, y32, x32, Mi, 768, 768);
        layernorm_kernel<<<Mi, 256, 0, stream>>>(y32, ln1_g + l * 768, ln1_b + l * 768, x32, x16);
        convert_transpose_kernel<<<ctF, 256, 0, stream>>>(w1 + wF, wt, 768, 3072);
        gemm_wmma_kernel<2><<<g3072, 256, 0, stream>>>(x16, wt, b1 + l * 3072, ff16, nullptr, nullptr, Mi, 3072, 768);
        convert_transpose_kernel<<<ctF, 256, 0, stream>>>(w2 + wF, wt, 3072, 768);
        gemm_wmma_kernel<3><<<g768, 256, 0, stream>>>(ff16, wt, b2 + l * 768, nullptr, y32, x32, Mi, 768, 3072);
        layernorm_kernel<<<Mi, 256, 0, stream>>>(y32, ln2_g + l * 768, ln2_b + l * 768, x32, x16);
    }

    // conv0 on WMMA (implicit GEMM, TDM-fed weight tile), then small head
    convert_conv0_kernel<<<3072, 256, 0, stream>>>(conv0_w, wt);
    conv0_wmma_kernel<<<dim3(1, 74), 256, 0, stream>>>(x16, wt, conv0_b,
                                                       bn0_g, bn0_b, bn0_m, bn0_v, c0);
    maxpool3_kernel<<<4480, 256, 0, stream>>>(c0, p0, 128, 37, 35);
    conv_bn_relu_kernel<<<896, 256, 0, stream>>>(p0, conv1_w, conv1_b, bn1_g, bn1_b, bn1_m, bn1_v,
                                                 c1, 128, 35, 64, 8, 2, 14);
    maxpool3_kernel<<<768, 256, 0, stream>>>(c1, p1, 64, 14, 12);
    conv_bn_relu_kernel<<<160, 256, 0, stream>>>(p1, conv2_w, conv2_b, bn2_g, bn2_b, bn2_m, bn2_v,
                                                 c2, 64, 12, 32, 3, 2, 5);
    maxpool3_kernel<<<96, 256, 0, stream>>>(c2, p2, 32, 5, 3);
    dense1_kernel<<<32, 256, 0, stream>>>(p2, d1_w, d1_b, fc1);
    dense2_kernel<<<1, 256, 0, stream>>>(fc1, d2_w, d2_b, out);
}
